// FlatGNN_9010841387609
// MI455X (gfx1250) — matrix operand order repaired and checked by reading, not compiled, not run on hardware
//
#include <hip/hip_runtime.h>
#include <hip/hip_bf16.h>

// ---------------------------------------------------------------------------
// Types for CDNA5 WMMA (gfx1250, wave32)
// ---------------------------------------------------------------------------
typedef __bf16 bf16;
typedef __attribute__((ext_vector_type(16))) __bf16 v16bf;
typedef __attribute__((ext_vector_type(8)))  __bf16 v8bf;
typedef __attribute__((ext_vector_type(8)))  float  v8f;

#define N_NODES 100000
#define D_IN    256
#define N_TOT   25600000     // N_NODES * D_IN
#define EPSLN   1e-5f

// workspace layout (floats):
//  [0..15]   scalars: 0=sum 1=sumsq 2=mu 3=rsigma 4=ssq_q 5=ssq_k 6=inv_nq 7=inv_nk
//  [16..271] ks_sum (256)
//  [272..16655] kvs raw f32 (4*64*64)
//  then bf16 regions: kvsB (16384), Wt (768*256), qbuf (N*256), kbuf (N*256)

static __device__ __forceinline__ v16bf cat8(v8bf a, v8bf b) {
    return __builtin_shufflevector(a, b, 0,1,2,3,4,5,6,7,8,9,10,11,12,13,14,15);
}

// ---------------------------------------------------------------------------
// 0) zero the accumulator region of ws
// ---------------------------------------------------------------------------
__global__ void init_ws(float* ws) {
    for (int i = threadIdx.x; i < 16656; i += 256) ws[i] = 0.0f;
}

// ---------------------------------------------------------------------------
// 1) global sum / sumsq of h_trans
// ---------------------------------------------------------------------------
__global__ __launch_bounds__(256) void reduce_stats(const float4* __restrict__ x,
                                                    float* __restrict__ s) {
    size_t i      = (size_t)blockIdx.x * blockDim.x + threadIdx.x;
    size_t stride = (size_t)gridDim.x * blockDim.x;
    const size_t n4 = N_TOT / 4;
    float sum = 0.f, ssq = 0.f;
    for (; i < n4; i += stride) {
        float4 v = x[i];
        sum += v.x + v.y + v.z + v.w;
        ssq += v.x*v.x + v.y*v.y + v.z*v.z + v.w*v.w;
    }
    for (int o = 16; o > 0; o >>= 1) {
        sum += __shfl_down(sum, o, 32);
        ssq += __shfl_down(ssq, o, 32);
    }
    __shared__ float ls[8], lq[8];
    int wid = threadIdx.x >> 5, lane = threadIdx.x & 31;
    if (lane == 0) { ls[wid] = sum; lq[wid] = ssq; }
    __syncthreads();
    if (threadIdx.x == 0) {
        float a = 0.f, b = 0.f;
        for (int w = 0; w < 8; ++w) { a += ls[w]; b += lq[w]; }
        atomicAdd(&s[0], a);
        atomicAdd(&s[1], b);
    }
}

__global__ void stats1(float* s) {
    if (threadIdx.x == 0 && blockIdx.x == 0) {
        const float n = (float)N_TOT;
        float mu  = s[0] / n;
        float var = s[1] / n - mu * mu;
        s[2] = mu;
        s[3] = rsqrtf(var + EPSLN);
    }
}

// ---------------------------------------------------------------------------
// 2) pack [Wq|Wk|Wv] transposed to bf16: Wt[j][k] = W(k, j%256), j in [0,768)
// ---------------------------------------------------------------------------
__global__ void pack_w(const float* __restrict__ Wq, const float* __restrict__ Wk,
                       const float* __restrict__ Wv, bf16* __restrict__ Wt) {
    int j = blockIdx.x;                                 // 0..767
    const float* W = (j < 256) ? Wq : ((j < 512) ? Wk : Wv);
    int jl = j & 255;
    for (int kk = threadIdx.x; kk < 256; kk += blockDim.x)
        Wt[(size_t)j * 256 + kk] = (bf16)W[(size_t)kk * 256 + jl];
}

// ---------------------------------------------------------------------------
// 3) fused LayerNorm + QKV projection, WMMA bf16 -> f32.
//    One block (8 waves) per 16-row tile. The normalized bf16 A slab is staged
//    in LDS ONCE, each wave caches its 8 A fragments in registers, then sweeps
//    6 of the 48 column tiles (Q:0..15, K:16..31, V:32..47) reloading only B.
// ---------------------------------------------------------------------------
__global__ __launch_bounds__(256) void qkv_wmma(
    const float* __restrict__ h, const bf16* __restrict__ Wt,
    const float* __restrict__ bq, const float* __restrict__ bk,
    const float* __restrict__ bv, const float* __restrict__ s,
    bf16* __restrict__ qbuf, bf16* __restrict__ kbuf, float* __restrict__ vout) {

    __shared__ bf16 A_lds[16 * 256];                    // normalized rows, bf16

    int t = threadIdx.x;
    int rt = blockIdx.x;                                // row tile 0..6249
    float mu = s[2], rs = s[3];

    // --- stage: each thread normalizes 16 contiguous elements of one row ---
    {
        int r    = t >> 4;                              // 0..15
        int cseg = (t & 15) << 4;                       // 0,16,...,240
        const float* src = h + (size_t)(rt * 16 + r) * D_IN + cseg;
        float4 f0 = *(const float4*)(src);
        float4 f1 = *(const float4*)(src + 4);
        float4 f2 = *(const float4*)(src + 8);
        float4 f3 = *(const float4*)(src + 12);
        v8bf o0, o1;
        o0[0] = (bf16)((f0.x - mu) * rs); o0[1] = (bf16)((f0.y - mu) * rs);
        o0[2] = (bf16)((f0.z - mu) * rs); o0[3] = (bf16)((f0.w - mu) * rs);
        o0[4] = (bf16)((f1.x - mu) * rs); o0[5] = (bf16)((f1.y - mu) * rs);
        o0[6] = (bf16)((f1.z - mu) * rs); o0[7] = (bf16)((f1.w - mu) * rs);
        o1[0] = (bf16)((f2.x - mu) * rs); o1[1] = (bf16)((f2.y - mu) * rs);
        o1[2] = (bf16)((f2.z - mu) * rs); o1[3] = (bf16)((f2.w - mu) * rs);
        o1[4] = (bf16)((f3.x - mu) * rs); o1[5] = (bf16)((f3.y - mu) * rs);
        o1[6] = (bf16)((f3.z - mu) * rs); o1[7] = (bf16)((f3.w - mu) * rs);
        *(v8bf*)(&A_lds[r * 256 + cseg])     = o0;
        *(v8bf*)(&A_lds[r * 256 + cseg + 8]) = o1;
    }
    __syncthreads();

    int w = t >> 5, lane = t & 31;
    int n16 = lane & 15, hi = lane >> 4;

    // --- A fragments for all 8 K-chunks, cached in registers ---
    v16bf Af[8];
    #pragma unroll
    for (int kc = 0; kc < 8; ++kc) {
        int kb = kc * 32;
        v8bf a0 = *(const v8bf*)(&A_lds[n16 * 256 + kb + hi * 8]);
        v8bf a1 = *(const v8bf*)(&A_lds[n16 * 256 + kb + 16 + hi * 8]);
        Af[kc] = cat8(a0, a1);
    }

    // --- sweep 6 column tiles per wave ---
    for (int j = 0; j < 6; ++j) {
        int ct = w * 6 + j;                             // 0..47, wave-uniform
        const bf16* brow = Wt + (size_t)(ct * 16 + n16) * 256;

        v8f acc = {0.f,0.f,0.f,0.f,0.f,0.f,0.f,0.f};
        #pragma unroll
        for (int kc = 0; kc < 8; ++kc) {
            int kb = kc * 32;
            v8bf b0 = *(const v8bf*)(brow + kb + hi * 8);
            v8bf b1 = *(const v8bf*)(brow + kb + 16 + hi * 8);
            acc = __builtin_amdgcn_wmma_f32_16x16x32_bf16(
                false, Af[kc], false, cat8(b0, b1), (short)0, acc, false, false);
        }

        int col  = ct * 16 + n16;                       // 0..767
        float bias = (col < 256) ? bq[col]
                   : ((col < 512) ? bk[col - 256] : bv[col - 512]);
        int rbase = rt * 16 + 8 * hi;                   // D row: i + 8*hi

        if (ct < 16) {
            #pragma unroll
            for (int i = 0; i < 8; ++i)
                qbuf[(size_t)(rbase + i) * 256 + col] = (bf16)(acc[i] + bias);
        } else if (ct < 32) {
            int c2 = col - 256;
            #pragma unroll
            for (int i = 0; i < 8; ++i)
                kbuf[(size_t)(rbase + i) * 256 + c2] = (bf16)(acc[i] + bias);
        } else {
            int c2 = col - 512;
            #pragma unroll
            for (int i = 0; i < 8; ++i)
                vout[(size_t)(rbase + i) * 256 + c2] = acc[i] + bias;
        }
    }
}

// ---------------------------------------------------------------------------
// 4) column stats over a bf16 [N,256] buffer: Frobenius sumsq (+ optional col sums)
// ---------------------------------------------------------------------------
__global__ __launch_bounds__(256) void col_stats(const bf16* __restrict__ buf,
                                                 float* __restrict__ ssq_out,
                                                 float* __restrict__ colsum_out) {
    int t = threadIdx.x;                                // column
    float ssq = 0.f, csum = 0.f;
    for (int r = blockIdx.x; r < N_NODES; r += gridDim.x) {
        float v = (float)buf[(size_t)r * 256 + t];
        ssq += v * v;
        csum += v;
    }
    if (colsum_out) atomicAdd(&colsum_out[t], csum);
    for (int o = 16; o > 0; o >>= 1) ssq += __shfl_down(ssq, o, 32);
    __shared__ float ls[8];
    int wid = t >> 5, lane = t & 31;
    if (lane == 0) ls[wid] = ssq;
    __syncthreads();
    if (t == 0) {
        float a = 0.f;
        for (int w = 0; w < 8; ++w) a += ls[w];
        atomicAdd(ssq_out, a);
    }
}

__global__ void stats2(float* s) {
    if (threadIdx.x == 0 && blockIdx.x == 0) {
        s[6] = rsqrtf(s[4]);   // 1 / ||q||_F
        s[7] = rsqrtf(s[5]);   // 1 / ||k||_F
    }
}

// ---------------------------------------------------------------------------
// 5) kvs[h][m][d] = sum_n k[n,h,m] * v[n,h,d]  (WMMA over LDS-transposed slabs)
//    Block = 8 waves: wave (half, h); each wave owns 2 m-tiles x 4 d-tiles.
// ---------------------------------------------------------------------------
__global__ __launch_bounds__(256) void kvs_wmma(const bf16* __restrict__ kbuf,
                                                const float* __restrict__ vsrc,
                                                float* __restrict__ kvs) {
    __shared__ bf16 kT[256 * 32];   // kT[col][n]
    __shared__ bf16 vT[256 * 32];   // vT[col][n]

    int t = threadIdx.x;
    int lane = t & 31, w = t >> 5;
    int hgl = w & 3;                // head
    int half = w >> 2;              // m-tile half
    int n16 = lane & 15, hi = lane >> 4;

    v8f acc[2][4];
    #pragma unroll
    for (int mt = 0; mt < 2; ++mt)
        #pragma unroll
        for (int dt = 0; dt < 4; ++dt)
            #pragma unroll
            for (int i = 0; i < 8; ++i) acc[mt][dt][i] = 0.f;

    int nstart = blockIdx.x * 512;
    for (int nb = 0; nb < 512; nb += 32) {
        // stage 32 rows of k,v into LDS transposed (bf16)
        for (int task = t; task < 1024; task += 256) {
            int r  = task >> 5;            // 0..31
            int cc = (task & 31) << 3;     // column base, 8 wide
            int gn = nstart + nb + r;
            if (gn < N_NODES) {
                v8bf kk = *(const v8bf*)(kbuf + (size_t)gn * 256 + cc);
                float4 v0 = *(const float4*)(vsrc + (size_t)gn * 256 + cc);
                float4 v1 = *(const float4*)(vsrc + (size_t)gn * 256 + cc + 4);
                #pragma unroll
                for (int e = 0; e < 8; ++e) kT[(cc + e) * 32 + r] = kk[e];
                vT[(cc + 0) * 32 + r] = (bf16)v0.x; vT[(cc + 1) * 32 + r] = (bf16)v0.y;
                vT[(cc + 2) * 32 + r] = (bf16)v0.z; vT[(cc + 3) * 32 + r] = (bf16)v0.w;
                vT[(cc + 4) * 32 + r] = (bf16)v1.x; vT[(cc + 5) * 32 + r] = (bf16)v1.y;
                vT[(cc + 6) * 32 + r] = (bf16)v1.z; vT[(cc + 7) * 32 + r] = (bf16)v1.w;
            } else {
                #pragma unroll
                for (int e = 0; e < 8; ++e) {
                    kT[(cc + e) * 32 + r] = (bf16)0.f;
                    vT[(cc + e) * 32 + r] = (bf16)0.f;
                }
            }
        }
        __syncthreads();

        v16bf Af[2];
        #pragma unroll
        for (int mt = 0; mt < 2; ++mt) {
            int m = hgl * 64 + (half * 2 + mt) * 16 + n16;
            v8bf a0 = *(const v8bf*)(&kT[m * 32 + hi * 8]);
            v8bf a1 = *(const v8bf*)(&kT[m * 32 + 16 + hi * 8]);
            Af[mt] = cat8(a0, a1);
        }
        #pragma unroll
        for (int dt = 0; dt < 4; ++dt) {
            int d = hgl * 64 + dt * 16 + n16;
            v8bf b0 = *(const v8bf*)(&vT[d * 32 + hi * 8]);
            v8bf b1 = *(const v8bf*)(&vT[d * 32 + 16 + hi * 8]);
            v16bf B = cat8(b0, b1);
            #pragma unroll
            for (int mt = 0; mt < 2; ++mt)
                acc[mt][dt] = __builtin_amdgcn_wmma_f32_16x16x32_bf16(
                    false, Af[mt], false, B, (short)0, acc[mt][dt], false, false);
        }
        __syncthreads();
    }

    #pragma unroll
    for (int mt = 0; mt < 2; ++mt)
        #pragma unroll
        for (int dt = 0; dt < 4; ++dt)
            #pragma unroll
            for (int i = 0; i < 8; ++i) {
                int m = (half * 2 + mt) * 16 + i + 8 * hi;
                int d = dt * 16 + n16;
                atomicAdd(&kvs[((size_t)hgl * 64 + m) * 64 + d], acc[mt][dt][i]);
            }
}

// ---------------------------------------------------------------------------
// 6) kvsB[h][d][m] = bf16( kvs[h][m][d] * inv_nq * inv_nk )
// ---------------------------------------------------------------------------
__global__ void pack_kvsb(const float* __restrict__ kvs, const float* __restrict__ s,
                          bf16* __restrict__ kvsB) {
    int idx = blockIdx.x * 256 + threadIdx.x;   // 0..16383
    float scale = s[6] * s[7];
    int hgl = idx >> 12;
    int rem = idx & 4095;
    int m = rem >> 6, d = rem & 63;
    kvsB[((size_t)hgl * 64 + d) * 64 + m] =
        (bf16)(kvs[((size_t)hgl * 64 + m) * 64 + d] * scale);
}

// ---------------------------------------------------------------------------
// 7) final: num = q_raw @ kvsB (pre-scaled) + N*v ; den = scale*(q.ks_sum) + N
//    One wave = one 16x16 tile of the [N,256] output; in-place over d_out.
//    Each lane computes the denominator for ONE row; shfl distributes them.
// ---------------------------------------------------------------------------
__global__ __launch_bounds__(256) void final_wmma(
    const bf16* __restrict__ qbuf, const bf16* __restrict__ kvsB,
    const float* __restrict__ kssum, const float* __restrict__ s,
    float* __restrict__ out) {

    __shared__ float lsk[256];
    lsk[threadIdx.x] = kssum[threadIdx.x];
    __syncthreads();

    int wave = (blockIdx.x << 3) + (threadIdx.x >> 5);  // 0..99999
    int lane = threadIdx.x & 31;
    int rt = wave >> 4;
    int ct = wave & 15;
    int hgl = ct >> 2, dt = ct & 3;
    int n16 = lane & 15, hi = lane >> 4;

    const bf16* qrow = qbuf + (size_t)(rt * 16 + n16) * 256 + hgl * 64;
    const bf16* brow = kvsB + (size_t)(hgl * 64 + dt * 16 + n16) * 64;

    v8f acc = {0.f,0.f,0.f,0.f,0.f,0.f,0.f,0.f};
    #pragma unroll
    for (int kc = 0; kc < 2; ++kc) {
        int kb = kc * 32;
        v8bf a0 = *(const v8bf*)(qrow + kb + hi * 8);
        v8bf a1 = *(const v8bf*)(qrow + kb + 16 + hi * 8);
        v8bf b0 = *(const v8bf*)(brow + kb + hi * 8);
        v8bf b1 = *(const v8bf*)(brow + kb + 16 + hi * 8);
        acc = __builtin_amdgcn_wmma_f32_16x16x32_bf16(
            false, cat8(a0, a1), false, cat8(b0, b1), (short)0, acc, false, false);
    }

    float scale = s[6] * s[7];
    const float Nf = (float)N_NODES;

    // den for row (rt*16 + n16), computed once per lane (lanes 16..31 mirror)
    float dot = 0.f;
    #pragma unroll
    for (int mo = 0; mo < 8; ++mo) {
        v8bf qv = *(const v8bf*)(qrow + mo * 8);        // same row as A fragment
        #pragma unroll
        for (int e = 0; e < 8; ++e)
            dot += (float)qv[e] * lsk[hgl * 64 + mo * 8 + e];
    }
    float den_mine = dot * scale + Nf;

    int colg  = hgl * 64 + dt * 16 + n16;
    int rbase = rt * 16 + 8 * hi;

    #pragma unroll
    for (int i = 0; i < 8; ++i) {
        int row = rbase + i;
        float den = __shfl(den_mine, i + 8 * hi, 32);   // row i+8*hi's den
        size_t o = (size_t)row * 256 + colg;
        float v = out[o];                               // v stored by qkv_wmma
        out[o] = (acc[i] + Nf * v) / den;
    }
}

// ---------------------------------------------------------------------------
// host launcher
// ---------------------------------------------------------------------------
extern "C" void kernel_launch(void* const* d_in, const int* in_sizes, int n_in,
                              void* d_out, int out_size, void* d_ws, size_t ws_size,
                              hipStream_t stream) {
    const float* h  = (const float*)d_in[0];
    const float* Wq = (const float*)d_in[1];
    const float* bq = (const float*)d_in[2];
    const float* Wk = (const float*)d_in[3];
    const float* bk = (const float*)d_in[4];
    const float* Wv = (const float*)d_in[5];
    const float* bv = (const float*)d_in[6];
    float* out = (float*)d_out;

    float* ws    = (float*)d_ws;
    float* s     = ws;            // 16 scalars
    float* kssum = ws + 16;       // 256
    float* kvs   = ws + 272;      // 16384
    bf16* kvsB = (bf16*)(ws + 16656);
    bf16* Wt   = kvsB + 16384;
    bf16* qbuf = Wt + (size_t)768 * 256;
    bf16* kbuf = qbuf + (size_t)N_NODES * 256;

    init_ws<<<1, 256, 0, stream>>>(ws);
    reduce_stats<<<2048, 256, 0, stream>>>((const float4*)h, s);
    stats1<<<1, 32, 0, stream>>>(s);
    pack_w<<<768, 256, 0, stream>>>(Wq, Wk, Wv, Wt);
    qkv_wmma<<<6250, 256, 0, stream>>>(h, Wt, bq, bk, bv, s, qbuf, kbuf, out);
    col_stats<<<400, 256, 0, stream>>>(qbuf, &s[4], nullptr);
    col_stats<<<400, 256, 0, stream>>>(kbuf, &s[5], kssum);
    stats2<<<1, 32, 0, stream>>>(s);
    kvs_wmma<<<196, 256, 0, stream>>>(kbuf, out, kvs);
    pack_kvsb<<<64, 256, 0, stream>>>(kvs, s, kvsB);
    final_wmma<<<12500, 256, 0, stream>>>(qbuf, kvsB, kssum, s, out);
}
